// MegrezMoeMoE_77412490543508
// MI455X (gfx1250) — compile-verified, matrix operand order checked by reading, not run on hardware
//
#include <hip/hip_runtime.h>
#include <hip/hip_bf16.h>

#define Hdim  2048
#define Edim  32
#define Idim  1024
#define TOPKn 6
#define Ttok  1024
#define SCALEF 2.5f

typedef __bf16 bf16_t;
typedef __attribute__((ext_vector_type(16))) bf16_t v16bf;
typedef __attribute__((ext_vector_type(8)))  bf16_t v8bf;
typedef __attribute__((ext_vector_type(8)))  float  v8f;
typedef __attribute__((ext_vector_type(4)))  int    v4i;

#if defined(__gfx1250__) && __has_builtin(__builtin_amdgcn_global_load_async_to_lds_b128) && __has_builtin(__builtin_amdgcn_s_wait_asynccnt)
#define USE_ASYNC_LDS 1
typedef __attribute__((address_space(1))) v4i gv4i;  // global AS pointee
typedef __attribute__((address_space(3))) v4i lv4i;  // LDS AS pointee
#else
#define USE_ASYNC_LDS 0
#endif

__device__ __forceinline__ v16bf cat8(v8bf lo, v8bf hi) {
  return __builtin_shufflevector(lo, hi, 0,1,2,3,4,5,6,7,8,9,10,11,12,13,14,15);
}

__device__ __forceinline__ void wave_argmax(float &v, int &idx) {
#pragma unroll
  for (int off = 16; off > 0; off >>= 1) {
    float ov = __shfl_xor(v, off, 32);
    int   oi = __shfl_xor(idx, off, 32);
    if (ov > v || (ov == v && oi < idx)) { v = ov; idx = oi; }
  }
}

// ---------------- routing ----------------

__global__ __launch_bounds__(32)
void init_kernel(int* cnt) { cnt[threadIdx.x & 31] = 0; }

__global__ __launch_bounds__(256)
void router_kernel(const float* __restrict__ X, const float* __restrict__ GW,
                   const float* __restrict__ bias,
                   int* __restrict__ topk_id, float* __restrict__ topk_w,
                   int* __restrict__ cnt) {
  const int lane = threadIdx.x & 31;
  const int t = blockIdx.x * 8 + (threadIdx.x >> 5);
  const float* x = X + (size_t)t * Hdim;

  float acc = 0.f;
  for (int h = 0; h < Hdim; h += 4) {
    float4 xv = *(const float4*)(x + h);
    acc += xv.x * GW[(h + 0) * Edim + lane];
    acc += xv.y * GW[(h + 1) * Edim + lane];
    acc += xv.z * GW[(h + 2) * Edim + lane];
    acc += xv.w * GW[(h + 3) * Edim + lane];
  }
  const float s  = 1.f / (1.f + __expf(-acc));   // sigmoid score
  const float sc = s + bias[lane];               // biased (selection only)

  // group score = top-2 sum within 4-lane group (8 groups)
  float b1  = __shfl_xor(sc, 1, 32);
  float mx  = fmaxf(sc, b1), mn = fminf(sc, b1);
  float mx2 = __shfl_xor(mx, 2, 32);
  float mn2 = __shfl_xor(mn, 2, 32);
  float gscore = fmaxf(mx, mx2) + fmaxf(fminf(mx, mx2), fmaxf(mn, mn2));

  // pick top-4 of 8 groups
  float gval = ((lane & 3) == 0) ? gscore : -3.0e38f;
  unsigned gsel = 0;
#pragma unroll
  for (int it = 0; it < 4; ++it) {
    float v = gval; int idx = lane;
    wave_argmax(v, idx);
    gsel |= 1u << (idx >> 2);
    if (lane == idx) gval = -3.0e38f;
  }
  float masked = ((gsel >> (lane >> 2)) & 1u) ? sc : -3.0e38f;

  int   wi[TOPKn]; float wsv[TOPKn]; float wsum = 0.f;
#pragma unroll
  for (int it = 0; it < TOPKn; ++it) {
    float v = masked; int idx = lane;
    wave_argmax(v, idx);
    wi[it]  = idx;
    float sw = __shfl(s, idx, 32);   // unbiased weight
    wsv[it] = sw; wsum += sw;
    if (lane == idx) masked = -3.0e38f;
  }
  if (lane == 0) {
    const float inv = 1.f / wsum;
#pragma unroll
    for (int i = 0; i < TOPKn; ++i) {
      topk_id[t * TOPKn + i] = wi[i];
      topk_w[t * TOPKn + i]  = wsv[i] * inv;
      atomicAdd(&cnt[wi[i]], 1);
    }
  }
}

__global__ __launch_bounds__(32)
void scan_kernel(const int* __restrict__ cnt, int* __restrict__ offs,
                 int* __restrict__ cursor) {
  const int lane = threadIdx.x & 31;
  int v = cnt[lane];
  int sum = v;
#pragma unroll
  for (int off = 1; off < 32; off <<= 1) {
    int n = __shfl_up(sum, off, 32);
    if (lane >= off) sum += n;
  }
  offs[lane]   = sum - v;
  cursor[lane] = sum - v;
}

__global__ __launch_bounds__(256)
void scatter_kernel(const int* __restrict__ topk_id, const float* __restrict__ topk_w,
                    int* __restrict__ cursor, int* __restrict__ rowtok,
                    float* __restrict__ roww) {
  const int t = blockIdx.x * 256 + threadIdx.x;
#pragma unroll
  for (int i = 0; i < TOPKn; ++i) {
    int e = topk_id[t * TOPKn + i];
    int pos = atomicAdd(&cursor[e], 1);
    rowtok[pos] = t;
    roww[pos]   = topk_w[t * TOPKn + i];
  }
}

// ---------------- GEMM 1: act = silu(X@Wg) * (X@Wu), bf16 out ----------------
// block tile 64(M) x 128(N of I), K step 32; 8 waves, wave tile 32x32

__global__ __launch_bounds__(256)
void gateup_kernel(const float* __restrict__ X, const float* __restrict__ Wall,
                   size_t wStride, const int* __restrict__ rowtok,
                   const int* __restrict__ cnts, const int* __restrict__ offs,
                   bf16_t* __restrict__ act, int Kdim, int ldW) {
  const int e   = blockIdx.z;
  const int cnt = cnts ? cnts[e] : Ttok;
  const int mBase = blockIdx.y * 64;
  if (mBase >= cnt) return;
  const int rowOff = offs ? offs[e] : 0;
  const int nBase  = blockIdx.x * 128;
  const float* W = Wall + (size_t)e * wStride;

  __shared__ __attribute__((aligned(16))) bf16_t sA [64][32];
  __shared__ __attribute__((aligned(16))) bf16_t sBg[128][32];
  __shared__ __attribute__((aligned(16))) bf16_t sBu[128][32];

  const int tid  = threadIdx.x;
  const int lane = tid & 31;
  const int wid  = tid >> 5;
  const int wm   = wid & 1;     // 2 M sub-tiles
  const int wn   = wid >> 1;    // 4 N sub-tiles

  // A staging
  const int ar = tid & 63;
  const int ak = (tid >> 6) * 8;
  const int gRowA = mBase + ar;
  const bool aval = gRowA < cnt;
  int tok = 0;
  if (aval) tok = rowtok ? rowtok[rowOff + gRowA] : gRowA;
  const float* xrow = X + (size_t)tok * Kdim;

  // B staging
  const int bn = tid & 127;
  const int bk = (tid >> 7) * 16;
  const int colg = nBase + bn;
  const int colu = colg + Idim;

  v8f accG[2][2] = {}; v8f accU[2][2] = {};
  const int fr = lane & 15;
  const int fk = (lane >> 4) * 8;

  for (int k0 = 0; k0 < Kdim; k0 += 32) {
    // ---- A tile (fp32 -> bf16) ----
    {
      float4 f0 = make_float4(0.f, 0.f, 0.f, 0.f), f1 = f0;
      if (aval) {
        f0 = *(const float4*)(xrow + k0 + ak);
        f1 = *(const float4*)(xrow + k0 + ak + 4);
      }
      v8bf av;
      av[0]=(bf16_t)f0.x; av[1]=(bf16_t)f0.y; av[2]=(bf16_t)f0.z; av[3]=(bf16_t)f0.w;
      av[4]=(bf16_t)f1.x; av[5]=(bf16_t)f1.y; av[6]=(bf16_t)f1.z; av[7]=(bf16_t)f1.w;
      *(v8bf*)&sA[ar][ak] = av;
    }
    // ---- B tiles (transpose into [n][k], fp32 -> bf16) ----
    {
      v8bf g0, g1, u0, u1;
#pragma unroll
      for (int kk = 0; kk < 8; ++kk) {
        const float* wr = W + (size_t)(k0 + bk + kk) * ldW;
        g0[kk] = (bf16_t)wr[colg];
        u0[kk] = (bf16_t)wr[colu];
      }
#pragma unroll
      for (int kk = 0; kk < 8; ++kk) {
        const float* wr = W + (size_t)(k0 + bk + 8 + kk) * ldW;
        g1[kk] = (bf16_t)wr[colg];
        u1[kk] = (bf16_t)wr[colu];
      }
      *(v8bf*)&sBg[bn][bk]     = g0;
      *(v8bf*)&sBg[bn][bk + 8] = g1;
      *(v8bf*)&sBu[bn][bk]     = u0;
      *(v8bf*)&sBu[bn][bk + 8] = u1;
      if (k0 + 32 < Kdim)  // pull next K-slab of weights toward L2
        __builtin_prefetch(W + (size_t)(k0 + 32 + bk) * ldW + colg, 0, 0);
    }
    __syncthreads();

    v16bf af[2], bg[2], bu[2];
#pragma unroll
    for (int i = 0; i < 2; ++i) {
      const bf16_t* p = &sA[wm * 32 + i * 16 + fr][fk];
      af[i] = cat8(*(const v8bf*)p, *(const v8bf*)(p + 16));
    }
#pragma unroll
    for (int j = 0; j < 2; ++j) {
      const bf16_t* pg = &sBg[wn * 32 + j * 16 + fr][fk];
      const bf16_t* pu = &sBu[wn * 32 + j * 16 + fr][fk];
      bg[j] = cat8(*(const v8bf*)pg, *(const v8bf*)(pg + 16));
      bu[j] = cat8(*(const v8bf*)pu, *(const v8bf*)(pu + 16));
    }
#pragma unroll
    for (int i = 0; i < 2; ++i)
#pragma unroll
      for (int j = 0; j < 2; ++j) {
        accG[i][j] = __builtin_amdgcn_wmma_f32_16x16x32_bf16(
            false, af[i], false, bg[j], (short)0, accG[i][j], false, false);
        accU[i][j] = __builtin_amdgcn_wmma_f32_16x16x32_bf16(
            false, af[i], false, bu[j], (short)0, accU[i][j], false, false);
      }
    __syncthreads();
  }

  // epilogue: silu(g)*u -> bf16 act
#pragma unroll
  for (int i = 0; i < 2; ++i)
#pragma unroll
    for (int j = 0; j < 2; ++j) {
      const int col = nBase + wn * 32 + j * 16 + (lane & 15);
#pragma unroll
      for (int r = 0; r < 8; ++r) {
        const int m  = wm * 32 + i * 16 + r + (lane >> 4) * 8;
        const int gR = mBase + m;
        if (gR < cnt) {
          float g = accG[i][j][r], u = accU[i][j][r];
          float a = (g / (1.f + __expf(-g))) * u;
          act[(size_t)(rowOff + gR) * Idim + col] = (bf16_t)a;
        }
      }
    }
}

// ---------------- GEMM 2: out = act @ Wd (store or scaled atomic add) --------

__global__ __launch_bounds__(256)
void down_kernel(const bf16_t* __restrict__ act, const float* __restrict__ Wall,
                 size_t wStride, const int* __restrict__ rowtok,
                 const float* __restrict__ roww, const int* __restrict__ cnts,
                 const int* __restrict__ offs, float* __restrict__ out, int mode) {
  const int e   = blockIdx.z;
  const int cnt = cnts ? cnts[e] : Ttok;
  const int mBase = blockIdx.y * 64;
  if (mBase >= cnt) return;
  const int rowOff = offs ? offs[e] : 0;
  const int nBase  = blockIdx.x * 128;
  const float* W = Wall + (size_t)e * wStride;

  __shared__ __attribute__((aligned(16))) bf16_t sA[64][32];
  __shared__ __attribute__((aligned(16))) bf16_t sB[128][32];

  const int tid  = threadIdx.x;
  const int lane = tid & 31;
  const int wid  = tid >> 5;
  const int wm   = wid & 1;
  const int wn   = wid >> 1;

  const int ar = tid & 63;
  const int ak = (tid >> 6) * 8;
  const int gRowA = mBase + ar;
  const bool aval = gRowA < cnt;
  const bf16_t* arow = act + (size_t)(rowOff + (aval ? gRowA : 0)) * Idim;

  const int bn = tid & 127;
  const int bk = (tid >> 7) * 16;
  const int coln = nBase + bn;

  v8f accD[2][2] = {};
  const int fr = lane & 15;
  const int fk = (lane >> 4) * 8;

  for (int k0 = 0; k0 < Idim; k0 += 32) {
    // ---- A tile: bf16 act, 16B/thread. Async DMA to LDS when available. ----
#if USE_ASYNC_LDS
    if (aval) {
      __builtin_amdgcn_global_load_async_to_lds_b128(
          (gv4i*)(arow + k0 + ak),
          (lv4i*)&sA[ar][ak],
          0, 0);
    } else {
      v8bf z = {};
      *(v8bf*)&sA[ar][ak] = z;
    }
#else
    {
      v8bf av = {};
      if (aval) av = *(const v8bf*)(arow + k0 + ak);
      *(v8bf*)&sA[ar][ak] = av;
    }
#endif
    // ---- B tile (transpose into [n][k], fp32 -> bf16) ----
    {
      v8bf b0, b1;
#pragma unroll
      for (int kk = 0; kk < 8; ++kk)
        b0[kk] = (bf16_t)W[(size_t)(k0 + bk + kk) * Hdim + coln];
#pragma unroll
      for (int kk = 0; kk < 8; ++kk)
        b1[kk] = (bf16_t)W[(size_t)(k0 + bk + 8 + kk) * Hdim + coln];
      *(v8bf*)&sB[bn][bk]     = b0;
      *(v8bf*)&sB[bn][bk + 8] = b1;
      if (k0 + 32 < Idim)
        __builtin_prefetch(W + (size_t)(k0 + 32 + bk) * Hdim + coln, 0, 0);
    }
#if USE_ASYNC_LDS
    __builtin_amdgcn_s_wait_asynccnt(0);
#endif
    __syncthreads();

    v16bf af[2], bf[2];
#pragma unroll
    for (int i = 0; i < 2; ++i) {
      const bf16_t* p = &sA[wm * 32 + i * 16 + fr][fk];
      af[i] = cat8(*(const v8bf*)p, *(const v8bf*)(p + 16));
    }
#pragma unroll
    for (int j = 0; j < 2; ++j) {
      const bf16_t* p = &sB[wn * 32 + j * 16 + fr][fk];
      bf[j] = cat8(*(const v8bf*)p, *(const v8bf*)(p + 16));
    }
#pragma unroll
    for (int i = 0; i < 2; ++i)
#pragma unroll
      for (int j = 0; j < 2; ++j)
        accD[i][j] = __builtin_amdgcn_wmma_f32_16x16x32_bf16(
            false, af[i], false, bf[j], (short)0, accD[i][j], false, false);
    __syncthreads();
  }

#pragma unroll
  for (int i = 0; i < 2; ++i)
#pragma unroll
    for (int j = 0; j < 2; ++j) {
      const int col = nBase + wn * 32 + j * 16 + (lane & 15);
#pragma unroll
      for (int r = 0; r < 8; ++r) {
        const int m  = wm * 32 + i * 16 + r + (lane >> 4) * 8;
        const int gR = mBase + m;
        if (gR < cnt) {
          float v = accD[i][j][r];
          if (mode) {
            int tok = rowtok[rowOff + gR];
            float w = roww[rowOff + gR];
            atomicAdd(&out[(size_t)tok * Hdim + col], SCALEF * w * v);
          } else {
            out[(size_t)gR * Hdim + col] = v;  // shared path: row == token
          }
        }
      }
    }
}

// ---------------- launch ----------------

extern "C" void kernel_launch(void* const* d_in, const int* in_sizes, int n_in,
                              void* d_out, int out_size, void* d_ws, size_t ws_size,
                              hipStream_t stream) {
  const float* X        = (const float*)d_in[0];
  const float* gate_w   = (const float*)d_in[1];
  const float* e_bias   = (const float*)d_in[2];
  const float* w_gu     = (const float*)d_in[3];
  const float* w_dn     = (const float*)d_in[4];
  const float* sh_gu    = (const float*)d_in[5];
  const float* sh_dn    = (const float*)d_in[6];
  float* out = (float*)d_out;

  // workspace carve-up (16B-aligned act region by construction)
  int*    cnt     = (int*)d_ws;                       // 32
  int*    offs    = cnt + 32;                         // 32
  int*    cursor  = offs + 32;                        // 32
  int*    topk_id = cursor + 32;                      // T*6
  float*  topk_w  = (float*)(topk_id + Ttok * TOPKn); // T*6
  int*    rowtok  = (int*)(topk_w + Ttok * TOPKn);    // T*6
  float*  roww    = (float*)(rowtok + Ttok * TOPKn);  // T*6
  bf16_t* act_r   = (bf16_t*)(roww + Ttok * TOPKn);   // (T*6) x I  bf16
  bf16_t* act_s   = act_r + (size_t)Ttok * TOPKn * Idim; // T x I bf16

  init_kernel<<<1, 32, 0, stream>>>(cnt);
  router_kernel<<<Ttok / 8, 256, 0, stream>>>(X, gate_w, e_bias, topk_id, topk_w, cnt);
  scan_kernel<<<1, 32, 0, stream>>>(cnt, offs, cursor);
  scatter_kernel<<<Ttok / 256, 256, 0, stream>>>(topk_id, topk_w, cursor, rowtok, roww);

  // shared expert: gate_up then down (plain stores initialize d_out)
  gateup_kernel<<<dim3(Idim / 128, Ttok / 64, 1), 256, 0, stream>>>(
      X, sh_gu, 0, nullptr, nullptr, nullptr, act_s, Hdim, 2 * Idim);
  down_kernel<<<dim3(Hdim / 128, Ttok / 64, 1), 256, 0, stream>>>(
      act_s, sh_dn, 0, nullptr, nullptr, nullptr, nullptr, out, 0);

  // routed experts: gathered GEMMs, scaled atomic accumulation
  gateup_kernel<<<dim3(Idim / 128, Ttok / 64, Edim), 256, 0, stream>>>(
      X, w_gu, (size_t)Hdim * 2 * Idim, rowtok, cnt, offs, act_r, Hdim, 2 * Idim);
  down_kernel<<<dim3(Hdim / 128, Ttok / 64, Edim), 256, 0, stream>>>(
      act_r, w_dn, (size_t)Idim * Hdim, rowtok, roww, cnt, offs, out, 1);
}
